// MultiScaleRepLK_76510547411175
// MI455X (gfx1250) — matrix operand order repaired and checked by reading, not compiled, not run on hardware
//
#include <hip/hip_runtime.h>

typedef __attribute__((ext_vector_type(16))) _Float16 v16h;
typedef __attribute__((ext_vector_type(8)))  _Float16 v8h;
typedef __attribute__((ext_vector_type(8)))  float    v8f;
typedef int v4i_ls __attribute__((vector_size(16)));   // matches async builtin param type

static constexpr int DIM = 256;
static constexpr int NB  = 16;    // batch
static constexpr int H   = 56;
static constexpr int W   = 56;
static constexpr int HP  = 62;    // padded height (halo 3)
static constexpr int WP  = 72;    // padded width  (halo 3 + tile slack)
static constexpr int BUF = 4 * 8 * 32 * 16;            // halfs per B stage buffer (32 KB)
static constexpr float EPS = 1e-5f;

#if defined(__has_builtin)
# if __has_builtin(__builtin_amdgcn_global_load_async_to_lds_b128)
#  define HAVE_ASYNC_LDS 1
# endif
#endif

// ---------------------------------------------------------------------------
// pack_x: NCHW f32 -> padded NHWC f16  xt[n][hp][wp][c], zero halo of 3.
// One thread handles 8 consecutive channels (16B contiguous store).
// ---------------------------------------------------------------------------
__global__ void pack_x(const float* __restrict__ x, _Float16* __restrict__ xt) {
  int idx = blockIdx.x * blockDim.x + threadIdx.x;          // NB*HP*WP*32
  if (idx >= NB * HP * WP * 32) return;
  int cg  = idx & 31;       int rem = idx >> 5;
  int wp  = rem % WP;       rem /= WP;
  int hp  = rem % HP;       int n = rem / HP;
  int h = hp - 3, w = wp - 3;
  v8h v = {};
  if (h >= 0 && h < H && w >= 0 && w < W) {
    int base = ((n * DIM + cg * 8) * H + h) * W + w;
#pragma unroll
    for (int j = 0; j < 8; ++j) v[j] = (_Float16)x[base + j * H * W];
  }
  *(v8h*)(xt + ((((long)n * HP + hp) * WP + wp) * DIM + cg * 8)) = v;
}

// ---------------------------------------------------------------------------
// pack_w: fold BN1/BN2 into conv weights (RepConv fusion) and emit WMMA
// A-fragments in per-lane order for V_WMMA_F32_16X16X32_F16.
// A 16x32 packing used here (contiguous k): lane<16 -> M=lane, K={0..15};
//                        lane>=16 -> M=lane-16, K={16..31}.
// Fragment index fr = (Ttile*9 + tap)*8 + kchunk ; 32 lanes x 16 halfs each.
// ---------------------------------------------------------------------------
__global__ void pack_w(const float* __restrict__ w3, const float* __restrict__ w1,
                       const float* __restrict__ g1, const float* __restrict__ b1,
                       const float* __restrict__ m1, const float* __restrict__ v1,
                       const float* __restrict__ g2, const float* __restrict__ b2,
                       const float* __restrict__ m2, const float* __restrict__ v2,
                       _Float16* __restrict__ A) {
  int idx = blockIdx.x * blockDim.x + threadIdx.x;          // 16*9*8*32
  if (idx >= 16 * 9 * 8 * 32) return;
  int lane = idx & 31; int fr = idx >> 5;
  int c = fr & 7; int rem = fr >> 3;
  int t = rem % 9; int T = rem / 9;
  int o = T * 16 + (lane & 15);
  int ky = t / 3, kx = t % 3;
  float s1 = g1[o] * rsqrtf(v1[o] + EPS);
  float s2 = g2[o] * rsqrtf(v2[o] + EPS);
  int base0 = (lane < 16) ? 0 : 8;
  v16h a;
#pragma unroll
  for (int j = 0; j < 16; ++j) {
    int k = 2 * base0 + j;                                  // contiguous: k = 16*(lane/16) + j
    int i = c * 32 + k;
    float wv = w3[((o * DIM + i) * 3 + ky) * 3 + kx] * s1;
    if (ky == 1 && kx == 1) wv += w1[o * DIM + i] * s2;
    a[j] = (_Float16)wv;
  }
  *(v16h*)(A + ((long)fr * 32 + lane) * 16) = a;
}

// ---------------------------------------------------------------------------
// prep_scalars: effective biases + BN3 affine + summed depthwise weights.
// ---------------------------------------------------------------------------
__global__ void prep_scalars(const float* g1, const float* b1, const float* m1, const float* v1,
                             const float* g2, const float* b2, const float* m2, const float* v2,
                             const float* g3, const float* b3, const float* m3, const float* v3,
                             const float* dw7, const float* dwk,
                             float* bias_eff, float* scale3, float* bias3b, float* dweff) {
  int o = blockIdx.x * blockDim.x + threadIdx.x;
  if (o >= DIM) return;
  float s1 = g1[o] * rsqrtf(v1[o] + EPS);
  float s2 = g2[o] * rsqrtf(v2[o] + EPS);
  bias_eff[o] = (b1[o] - m1[o] * s1) + (b2[o] - m2[o] * s2);
  float s3 = g3[o] * rsqrtf(v3[o] + EPS);
  scale3[o] = s3;
  bias3b[o] = b3[o] - m3[o] * s3;
  for (int j = 0; j < 49; ++j) dweff[o * 49 + j] = dw7[o * 49 + j] + dwk[o * 49 + j];
}

// ---------------------------------------------------------------------------
// dwconv: fused 7x7 depthwise (dw7+dwk), coalesced along w, f32.
// ---------------------------------------------------------------------------
__global__ void dwconv(const float* __restrict__ x, const float* __restrict__ dweff,
                       float* __restrict__ dwout) {
  int idx = blockIdx.x * blockDim.x + threadIdx.x;
  if (idx >= NB * DIM * H * W) return;
  int w = idx % W; int rem = idx / W;
  int h = rem % H; rem /= H;
  int c = rem % DIM; int n = rem / DIM;
  const float* xp = x + ((long)(n * DIM + c) * H) * W;
  const float* wp = dweff + c * 49;
  float acc = 0.f;
#pragma unroll
  for (int dy = -3; dy <= 3; ++dy) {
    int hh = h + dy;
    if (hh < 0 || hh >= H) continue;
#pragma unroll
    for (int dx = -3; dx <= 3; ++dx) {
      int ww = w + dx;
      if (ww < 0 || ww >= W) continue;
      acc += xp[hh * W + ww] * wp[(dy + 3) * 7 + (dx + 3)];
    }
  }
  dwout[idx] = acc;
}

// ---------------------------------------------------------------------------
// fused_main: implicit-GEMM 3x3 conv via WMMA + fused epilogue.
// Block = 512 threads = 16 wave32s; wave T owns cout tile [16T, 16T+16).
// One block per (n, h) output ROW: 4 spatial tiles of 16 (W=56, tile 3 masked
// on store; padded xt makes halo reads return 0). Each wave keeps 4
// accumulators and reuses every A fragment register 4x.
// K loop: 9 taps x 8 chunks of 32 cin -> 288 v_wmma_f32_16x16x32_f16 / wave.
// B is DOUBLE-BUFFERED through 2 x 32 KB of LDS via ASYNCcnt-tracked
// GLOBAL_LOAD_ASYNC_TO_LDS_B128: tap t+1's DMA overlaps tap t's WMMAs.
// ---------------------------------------------------------------------------
__global__ __launch_bounds__(512)
void fused_main(const _Float16* __restrict__ xt, const _Float16* __restrict__ A,
                const float* __restrict__ bias_eff, const float* __restrict__ scale3,
                const float* __restrict__ bias3b, const float* __restrict__ dwout,
                float* __restrict__ out) {
  __shared__ alignas(32) _Float16 Bsh[2 * BUF];        // 64 KB
  int tid  = threadIdx.x;
  int lane = tid & 31;
  int wave = tid >> 5;                                 // cout tile T
  int h = blockIdx.x % H;
  int n = blockIdx.x / H;

  // stage tap t's B fragments (4 tiles x 8 chunks x 32 lanes = 1024 32B jobs)
  auto stage = [&](int buf, int t) {
    int dy = t / 3, dx = t % 3;
#pragma unroll
    for (int j = tid; j < 1024; j += 512) {
      int tile = j >> 8;
      int c    = (j >> 5) & 7;
      int L    = j & 31;
      int row = h + dy + 2;                            // (h+3) + (dy-1)
      int col = tile * 16 + (L & 15) + dx + 2;         // (w+3) + (dx-1)
      const _Float16* src = xt + ((((long)n * HP + row) * WP + col) * DIM
                                  + c * 32 + ((L & 16) ? 16 : 0));
      _Float16* dst = Bsh + buf * BUF + ((tile * 8 + c) * 32 + L) * 16;
#if HAVE_ASYNC_LDS
      __builtin_amdgcn_global_load_async_to_lds_b128(
          (__attribute__((address_space(1))) v4i_ls*)src,
          (__attribute__((address_space(3))) v4i_ls*)dst, 0, 0);
      __builtin_amdgcn_global_load_async_to_lds_b128(
          (__attribute__((address_space(1))) v4i_ls*)(src + 8),
          (__attribute__((address_space(3))) v4i_ls*)(dst + 8), 0, 0);
#else
      *(v16h*)dst = *(const v16h*)src;
#endif
    }
  };

  v8f acc[4] = {{}, {}, {}, {}};
  const _Float16* Abase = A + ((long)(wave * 9) * 8 * 32 + lane) * 16;

  stage(0, 0);                                         // prologue: fill buffer 0
#if HAVE_ASYNC_LDS
  asm volatile("s_wait_asynccnt 0" ::: "memory");
#endif
  __syncthreads();

  for (int t = 0; t < 9; ++t) {
    int cur = t & 1;
    if (t < 8) stage(1 - cur, t + 1);                  // overlap next tap's DMA
    const _Float16* Ap = Abase + (long)t * 8 * 512;    // fragment stride = 32*16 halfs
    const _Float16* Bp = Bsh + cur * BUF;
    if (t < 8) __builtin_prefetch((const void*)(Ap + 8L * 512), 0, 3);
#pragma unroll
    for (int c = 0; c < 8; ++c) {
      v16h a = *(const v16h*)(Ap + (long)c * 512);
#pragma unroll
      for (int ti = 0; ti < 4; ++ti) {
        v16h b = *(const v16h*)(Bp + ((ti * 8 + c) * 32 + lane) * 16);
        acc[ti] = __builtin_amdgcn_wmma_f32_16x16x32_f16(false, a, false, b,
                                                         (short)0, acc[ti], false, false);
      }
    }
    if (t < 8) {
#if HAVE_ASYNC_LDS
      asm volatile("s_wait_asynccnt 0" ::: "memory");
#endif
      __syncthreads();                                 // next buf visible; cur buf free
    }
  }

  // Epilogue. 16x16 f32 D layout: VGPR r -> M = r (lanes 0-15) / r+8 (lanes 16-31),
  // N = lane & 15 (spatial column).
  int lw    = lane & 15;
  int obase = wave * 16 + ((lane & 16) ? 8 : 0);
#pragma unroll
  for (int ti = 0; ti < 4; ++ti) {
    int wcol = ti * 16 + lw;
    if (wcol < W) {
#pragma unroll
      for (int r = 0; r < 8; ++r) {
        int o = obase + r;
        float z   = acc[ti][r] + bias_eff[o];
        float rep = z / (1.f + __expf(-z));            // SiLU
        int oi = ((n * DIM + o) * H + h) * W + wcol;
        float y  = rep + dwout[oi];
        float zz = y * scale3[o] + bias3b[o];
        out[oi]  = zz / (1.f + __expf(-zz));           // SiLU
      }
    }
  }
}

// ---------------------------------------------------------------------------
// host launcher
// ---------------------------------------------------------------------------
extern "C" void kernel_launch(void* const* d_in, const int* in_sizes, int n_in,
                              void* d_out, int out_size, void* d_ws, size_t ws_size,
                              hipStream_t stream) {
  (void)in_sizes; (void)n_in; (void)out_size; (void)ws_size;
  const float* x   = (const float*)d_in[0];
  const float* w3  = (const float*)d_in[1];
  const float* g1  = (const float*)d_in[2];
  const float* b1  = (const float*)d_in[3];
  const float* m1  = (const float*)d_in[4];
  const float* v1  = (const float*)d_in[5];
  const float* w1  = (const float*)d_in[6];
  const float* g2  = (const float*)d_in[7];
  const float* b2  = (const float*)d_in[8];
  const float* m2  = (const float*)d_in[9];
  const float* v2  = (const float*)d_in[10];
  const float* dw7 = (const float*)d_in[11];
  const float* dwk = (const float*)d_in[12];
  const float* g3  = (const float*)d_in[13];
  const float* b3  = (const float*)d_in[14];
  const float* m3  = (const float*)d_in[15];
  const float* v3  = (const float*)d_in[16];

  // workspace layout (bytes), ~89.2 MB total
  char* ws = (char*)d_ws;
  _Float16* xt    = (_Float16*)(ws);                    // 16*62*72*256*2 = 36,569,088
  _Float16* Afr   = (_Float16*)(ws + 36569088);         // 16*9*8*32*16*2 =  1,179,648
  float*    dwout = (float*)   (ws + 37748736);         // 16*256*56*56*4 = 51,380,224
  float*    bias_eff = (float*)(ws + 89128960);         // 256 f32
  float*    scale3   = bias_eff + 256;
  float*    bias3b   = scale3 + 256;
  float*    dweff    = bias3b + 256;                    // 256*49 f32

  {
    int total = NB * HP * WP * 32;
    pack_x<<<(total + 255) / 256, 256, 0, stream>>>(x, xt);
  }
  pack_w<<<(16 * 9 * 8 * 32 + 255) / 256, 256, 0, stream>>>(
      w3, w1, g1, b1, m1, v1, g2, b2, m2, v2, Afr);
  prep_scalars<<<1, 256, 0, stream>>>(g1, b1, m1, v1, g2, b2, m2, v2,
                                      g3, b3, m3, v3, dw7, dwk,
                                      bias_eff, scale3, bias3b, dweff);
  {
    int total = NB * DIM * H * W;
    dwconv<<<(total + 255) / 256, 256, 0, stream>>>(x, dweff, dwout);
  }
  fused_main<<<NB * H, 512, 0, stream>>>(xt, Afr, bias_eff, scale3, bias3b,
                                         dwout, (float*)d_out);
}
